// SplitSeek_50251117363665
// MI455X (gfx1250) — matrix operand order repaired and checked by reading, not compiled, not run on hardware
//
#include <hip/hip_runtime.h>
#include <math.h>

// ---------------------------------------------------------------------------
// MPNN layer for MI455X (gfx1250): bf16 WMMA (v_wmma_f32_16x16x32_bf16),
// f32 accumulate, LDS-staged A operands, weights streamed from L2 in
// pre-swizzled B-fragment order.
// ---------------------------------------------------------------------------

namespace {

constexpr int HD   = 128;   // hidden
constexpr int NN   = 2048;  // nodes per batch
constexpr int BB   = 2;
constexpr int KK   = 48;    // neighbors
constexpr int NNOD = BB * NN;  // 4096 flat nodes
constexpr float SCALE_INV = 1.0f / 30.0f;
constexpr float LN_EPS = 1e-5f;

constexpr int A_STRIDE = 392;  // 384 + 8 pad (ushort units, keeps 16B align)
constexpr int H_STRIDE = 136;  // 128 + 8
constexpr int F_STRIDE = 520;  // 512 + 8
constexpr int E_STRIDE = 132;  // f32 row stride for edge LN dump

typedef __attribute__((ext_vector_type(8)))  unsigned short v8us;
typedef __attribute__((ext_vector_type(16))) unsigned short v16us;
typedef __attribute__((ext_vector_type(16))) __bf16        v16bf;
typedef __attribute__((ext_vector_type(8)))  float         v8f;

union FragU { v16us u; v8us h[2]; };

__device__ __forceinline__ unsigned short f2bf(float f) {
  unsigned u = __float_as_uint(f);
  return (unsigned short)((u + 0x7FFFu + ((u >> 16) & 1u)) >> 16);  // RNE
}
__device__ __forceinline__ float gelu_erf(float x) {
  return 0.5f * x * (1.0f + erff(x * 0.70710678118654752f));
}
__device__ __forceinline__ v8f vzero() {
  v8f r;
#pragma unroll
  for (int i = 0; i < 8; ++i) r[i] = 0.0f;
  return r;
}

// A fragment (16x32 bf16) from row-major bf16 LDS tile.
// lane<16: M=lane, K = k0+0..7 and k0+16..23 ; lane>=16: M=lane-16, K = k0+8..15, k0+24..31
__device__ __forceinline__ v16us ldsA(const unsigned short* base, int strideU,
                                      int row0, int k0, int lane) {
  const int m = lane & 15, hi = lane >> 4;
  const unsigned short* p = base + (row0 + m) * strideU + k0 + hi * 8;
  FragU f;
  f.h[0] = *(const v8us*)p;         // K +0..7
  f.h[1] = *(const v8us*)(p + 16);  // K +16..23
  return f.u;
}
// B fragment (32x16 bf16) from pre-swizzled weights: 32 lanes x 16 bf16 per chunk.
__device__ __forceinline__ v16us ldw(const unsigned short* __restrict__ wp,
                                     int tile, int lane) {
  const unsigned short* p = wp + (((unsigned)tile * 32u + (unsigned)lane) << 4);
  FragU f;
  f.h[0] = *(const v8us*)p;
  f.h[1] = *(const v8us*)(p + 8);
  return f.u;
}
__device__ __forceinline__ v8f wmma_bf(v16us a, v16us b, v8f c) {
  return __builtin_amdgcn_wmma_f32_16x16x32_bf16(
      false, __builtin_bit_cast(v16bf, a),
      false, __builtin_bit_cast(v16bf, b),
      (short)0, c, false, false);
}
__device__ __forceinline__ float wave_sum(float s) {
#pragma unroll
  for (int o = 16; o >= 1; o >>= 1) s += __shfl_xor(s, o, 32);
  return s;
}

}  // namespace

// ---------------------------------------------------------------------------
// Pack weight [Kdim x Ncols] f32 -> B-fragment-ordered bf16:
// dst[((t*CH + c)*32 + lane)*16 + j] = bf16(src[(c*32 + (lane>>4)*8 + (j&7) + (j>>3)*16)*Ncols + t*16 + (lane&15)])
// ---------------------------------------------------------------------------
__global__ void __launch_bounds__(256) pack_w_kernel(const float* __restrict__ src,
                                                     unsigned short* __restrict__ dst,
                                                     int Kdim, int Ncols) {
  int i = blockIdx.x * 256 + threadIdx.x;
  if (i >= Kdim * Ncols) return;
  int j = i & 15, l = (i >> 4) & 31, rest = i >> 9;
  int CH = Kdim >> 5;
  int c = rest % CH, t = rest / CH;
  int n = t * 16 + (l & 15);
  int k = c * 32 + (l >> 4) * 8 + (j & 7) + (j >> 3) * 16;
  dst[i] = f2bf(src[k * Ncols + n]);
}

// ---------------------------------------------------------------------------
// Kernel 1: node message pass. One workgroup per node; 8 waves, each owns a
// 16-column strip; 3 M-tiles of 16 edge-rows. Writes dh into out-hVn region.
// ---------------------------------------------------------------------------
__global__ void __launch_bounds__(256) node_msg_kernel(
    const float* __restrict__ hV, const float* __restrict__ hE,
    const int* __restrict__ Eidx, const float* __restrict__ maskA,
    const unsigned short* __restrict__ W1p, const unsigned short* __restrict__ W2p,
    const unsigned short* __restrict__ W3p,
    const float* __restrict__ b1, const float* __restrict__ b2,
    const float* __restrict__ b3, float* __restrict__ dh) {
  __shared__ __align__(16) unsigned short As[KK * A_STRIDE];
  __shared__ __align__(16) unsigned short Hs0[KK * H_STRIDE];
  __shared__ __align__(16) unsigned short Hs1[KK * H_STRIDE];
  __shared__ float mrow[KK];

  const int nf = blockIdx.x;  // flat node 0..B*N-1
  const int tid = threadIdx.x;
  const int lane = tid & 31, wave = tid >> 5;
  const int hi = lane >> 4;
  const int col = wave * 16 + (lane & 15);

  // Stage A = [hV | hE | gather(hV)] rows as bf16
  const float* vrow = hV + (long)nf * HD;
  const int b = nf >> 11;  // N == 2048
  for (int i = tid; i < KK * 384; i += 256) {
    int r = i / 384, c = i - r * 384;
    float v;
    if (c < 128) {
      v = vrow[c];
    } else if (c < 256) {
      v = hE[((long)nf * KK + r) * HD + (c - 128)];
    } else {
      int g = Eidx[(long)nf * KK + r];
      v = hV[((long)(b * NN + g)) * HD + (c - 256)];
    }
    As[r * A_STRIDE + c] = f2bf(v);
  }
  if (tid < KK) mrow[tid] = maskA[(long)nf * KK + tid];
  __syncthreads();

  v8f acc[3];
#pragma unroll
  for (int mt = 0; mt < 3; ++mt) acc[mt] = vzero();

  // GEMM1: 48x384 @ 384x128
  for (int c = 0; c < 12; ++c) {
    v16us bfr = ldw(W1p, wave * 12 + c, lane);
#pragma unroll
    for (int mt = 0; mt < 3; ++mt) {
      v16us af = ldsA(As, A_STRIDE, mt * 16, c * 32, lane);
      acc[mt] = wmma_bf(af, bfr, acc[mt]);
    }
  }
  {
    float bias = b1[col];
#pragma unroll
    for (int mt = 0; mt < 3; ++mt)
#pragma unroll
      for (int v = 0; v < 8; ++v)
        Hs0[(mt * 16 + hi * 8 + v) * H_STRIDE + col] =
            f2bf(gelu_erf(acc[mt][v] + bias));
  }
  __syncthreads();

  // GEMM2: 48x128 @ 128x128
#pragma unroll
  for (int mt = 0; mt < 3; ++mt) acc[mt] = vzero();
  for (int c = 0; c < 4; ++c) {
    v16us bfr = ldw(W2p, wave * 4 + c, lane);
#pragma unroll
    for (int mt = 0; mt < 3; ++mt) {
      v16us af = ldsA(Hs0, H_STRIDE, mt * 16, c * 32, lane);
      acc[mt] = wmma_bf(af, bfr, acc[mt]);
    }
  }
  {
    float bias = b2[col];
#pragma unroll
    for (int mt = 0; mt < 3; ++mt)
#pragma unroll
      for (int v = 0; v < 8; ++v)
        Hs1[(mt * 16 + hi * 8 + v) * H_STRIDE + col] =
            f2bf(gelu_erf(acc[mt][v] + bias));
  }
  __syncthreads();

  // GEMM3: 48x128 @ 128x128
#pragma unroll
  for (int mt = 0; mt < 3; ++mt) acc[mt] = vzero();
  for (int c = 0; c < 4; ++c) {
    v16us bfr = ldw(W3p, wave * 4 + c, lane);
#pragma unroll
    for (int mt = 0; mt < 3; ++mt) {
      v16us af = ldsA(Hs1, H_STRIDE, mt * 16, c * 32, lane);
      acc[mt] = wmma_bf(af, bfr, acc[mt]);
    }
  }
  // masked row-reduction over K=48, write dh/SCALE
  {
    float bias = b3[col];
    float tot = 0.0f;
#pragma unroll
    for (int mt = 0; mt < 3; ++mt)
#pragma unroll
      for (int v = 0; v < 8; ++v) {
        int r = mt * 16 + hi * 8 + v;
        tot += mrow[r] * (acc[mt][v] + bias);
      }
    tot += __shfl_xor(tot, 16, 32);
    if (lane < 16) dh[(long)nf * HD + col] = tot * SCALE_INV;
  }
}

// ---------------------------------------------------------------------------
// Kernel 2: x = hV + dh ; ln1 ; FFN(128->512->128 GELU) ; ln2 ; * mask_V.
// 16 nodes per workgroup.
// ---------------------------------------------------------------------------
__global__ void __launch_bounds__(256) node_ffn_kernel(
    const float* __restrict__ hV, const float* dhin,
    const unsigned short* __restrict__ Winp, const unsigned short* __restrict__ Woutp,
    const float* __restrict__ bin, const float* __restrict__ bout,
    const float* __restrict__ ln1g, const float* __restrict__ ln1b,
    const float* __restrict__ ln2g, const float* __restrict__ ln2b,
    const float* __restrict__ maskV, float* outV) {
  __shared__ float Xf[16 * HD];
  __shared__ __align__(16) unsigned short Xs[16 * H_STRIDE];
  __shared__ __align__(16) unsigned short Hs[16 * F_STRIDE];

  const int g = blockIdx.x;
  const int tid = threadIdx.x;
  const int lane = tid & 31, wave = tid >> 5;
  const int hi = lane >> 4;
  const long base = (long)g * 16 * HD;

  for (int i = tid; i < 16 * HD; i += 256) Xf[i] = hV[base + i] + dhin[base + i];
  __syncthreads();

  // LN1 (2 rows per wave), write f32 back + bf16 copy for GEMM A
#pragma unroll
  for (int rr = 0; rr < 2; ++rr) {
    int r = wave * 2 + rr;
    float x[4];
    float s = 0.0f;
#pragma unroll
    for (int q = 0; q < 4; ++q) { x[q] = Xf[r * HD + lane + q * 32]; s += x[q]; }
    float m = wave_sum(s) * (1.0f / 128.0f);
    float vs = 0.0f;
#pragma unroll
    for (int q = 0; q < 4; ++q) { float d = x[q] - m; vs += d * d; }
    float rstd = rsqrtf(wave_sum(vs) * (1.0f / 128.0f) + LN_EPS);
#pragma unroll
    for (int q = 0; q < 4; ++q) {
      int cc = lane + q * 32;
      float ln = (x[q] - m) * rstd * ln1g[cc] + ln1b[cc];
      Xf[r * HD + cc] = ln;
      Xs[r * H_STRIDE + cc] = f2bf(ln);
    }
  }
  __syncthreads();

  // GEMM1: 16x128 @ 128x512 ; GELU -> Hs (bf16)
  v16us afr[4];
#pragma unroll
  for (int c = 0; c < 4; ++c) afr[c] = ldsA(Xs, H_STRIDE, 0, c * 32, lane);
  for (int nt4 = 0; nt4 < 4; ++nt4) {
    int nt = wave * 4 + nt4;
    v8f acc = vzero();
#pragma unroll
    for (int c = 0; c < 4; ++c) acc = wmma_bf(afr[c], ldw(Winp, nt * 4 + c, lane), acc);
    int cw = nt * 16 + (lane & 15);
    float bias = bin[cw];
#pragma unroll
    for (int v = 0; v < 8; ++v)
      Hs[(hi * 8 + v) * F_STRIDE + cw] = f2bf(gelu_erf(acc[v] + bias));
  }
  __syncthreads();

  // GEMM2: 16x512 @ 512x128 ; residual add into Xf
  {
    v8f acc = vzero();
    for (int c = 0; c < 16; ++c)
      acc = wmma_bf(ldsA(Hs, F_STRIDE, 0, c * 32, lane),
                    ldw(Woutp, wave * 16 + c, lane), acc);
    int co = wave * 16 + (lane & 15);
    float bias = bout[co];
#pragma unroll
    for (int v = 0; v < 8; ++v) {
      int r = hi * 8 + v;
      Xf[r * HD + co] += acc[v] + bias;
    }
  }
  __syncthreads();

  // LN2 + mask_V -> out
#pragma unroll
  for (int rr = 0; rr < 2; ++rr) {
    int r = wave * 2 + rr;
    float x[4];
    float s = 0.0f;
#pragma unroll
    for (int q = 0; q < 4; ++q) { x[q] = Xf[r * HD + lane + q * 32]; s += x[q]; }
    float m = wave_sum(s) * (1.0f / 128.0f);
    float vs = 0.0f;
#pragma unroll
    for (int q = 0; q < 4; ++q) { float d = x[q] - m; vs += d * d; }
    float rstd = rsqrtf(wave_sum(vs) * (1.0f / 128.0f) + LN_EPS);
    float mk = maskV[g * 16 + r];
#pragma unroll
    for (int q = 0; q < 4; ++q) {
      int cc = lane + q * 32;
      outV[base + r * HD + cc] = mk * ((x[q] - m) * rstd * ln2g[cc] + ln2b[cc]);
    }
  }
}

// ---------------------------------------------------------------------------
// Kernel 3: edge update. Same MLP structure on [hVn | hE | gather(hVn)],
// residual + LN3, write h_En.
// ---------------------------------------------------------------------------
__global__ void __launch_bounds__(256) edge_upd_kernel(
    const float* __restrict__ hVn, const float* __restrict__ hE,
    const int* __restrict__ Eidx,
    const unsigned short* __restrict__ W11p, const unsigned short* __restrict__ W12p,
    const unsigned short* __restrict__ W13p,
    const float* __restrict__ b11, const float* __restrict__ b12,
    const float* __restrict__ b13,
    const float* __restrict__ ln3g, const float* __restrict__ ln3b,
    float* __restrict__ outE) {
  __shared__ __align__(16) unsigned short As[KK * A_STRIDE];
  __shared__ __align__(16) unsigned short Hs0[KK * H_STRIDE];
  __shared__ __align__(16) unsigned short Hs1[KK * H_STRIDE];

  const int nf = blockIdx.x;
  const int tid = threadIdx.x;
  const int lane = tid & 31, wave = tid >> 5;
  const int hi = lane >> 4;
  const int col = wave * 16 + (lane & 15);

  const float* vrow = hVn + (long)nf * HD;
  const int b = nf >> 11;
  for (int i = tid; i < KK * 384; i += 256) {
    int r = i / 384, c = i - r * 384;
    float v;
    if (c < 128) {
      v = vrow[c];
    } else if (c < 256) {
      v = hE[((long)nf * KK + r) * HD + (c - 128)];
    } else {
      int g = Eidx[(long)nf * KK + r];
      v = hVn[((long)(b * NN + g)) * HD + (c - 256)];
    }
    As[r * A_STRIDE + c] = f2bf(v);
  }
  __syncthreads();

  v8f acc[3];
#pragma unroll
  for (int mt = 0; mt < 3; ++mt) acc[mt] = vzero();
  for (int c = 0; c < 12; ++c) {
    v16us bfr = ldw(W11p, wave * 12 + c, lane);
#pragma unroll
    for (int mt = 0; mt < 3; ++mt) {
      v16us af = ldsA(As, A_STRIDE, mt * 16, c * 32, lane);
      acc[mt] = wmma_bf(af, bfr, acc[mt]);
    }
  }
  {
    float bias = b11[col];
#pragma unroll
    for (int mt = 0; mt < 3; ++mt)
#pragma unroll
      for (int v = 0; v < 8; ++v)
        Hs0[(mt * 16 + hi * 8 + v) * H_STRIDE + col] =
            f2bf(gelu_erf(acc[mt][v] + bias));
  }
  __syncthreads();

#pragma unroll
  for (int mt = 0; mt < 3; ++mt) acc[mt] = vzero();
  for (int c = 0; c < 4; ++c) {
    v16us bfr = ldw(W12p, wave * 4 + c, lane);
#pragma unroll
    for (int mt = 0; mt < 3; ++mt) {
      v16us af = ldsA(Hs0, H_STRIDE, mt * 16, c * 32, lane);
      acc[mt] = wmma_bf(af, bfr, acc[mt]);
    }
  }
  {
    float bias = b12[col];
#pragma unroll
    for (int mt = 0; mt < 3; ++mt)
#pragma unroll
      for (int v = 0; v < 8; ++v)
        Hs1[(mt * 16 + hi * 8 + v) * H_STRIDE + col] =
            f2bf(gelu_erf(acc[mt][v] + bias));
  }
  __syncthreads();

#pragma unroll
  for (int mt = 0; mt < 3; ++mt) acc[mt] = vzero();
  for (int c = 0; c < 4; ++c) {
    v16us bfr = ldw(W13p, wave * 4 + c, lane);
#pragma unroll
    for (int mt = 0; mt < 3; ++mt) {
      v16us af = ldsA(Hs1, H_STRIDE, mt * 16, c * 32, lane);
      acc[mt] = wmma_bf(af, bfr, acc[mt]);
    }
  }

  // residual + dump f32 rows (reuse dead A-stage LDS; all waves passed the
  // barrier after GEMM1, so As is no longer read)
  float* Es = (float*)As;  // 48 * E_STRIDE floats = 25344 B <= 37632 B
  {
    float bias = b13[col];
#pragma unroll
    for (int mt = 0; mt < 3; ++mt)
#pragma unroll
      for (int v = 0; v < 8; ++v) {
        int r = mt * 16 + hi * 8 + v;
        Es[r * E_STRIDE + col] =
            acc[mt][v] + bias + hE[((long)nf * KK + r) * HD + col];
      }
  }
  __syncthreads();

  // LN3 per edge row, write h_En
  for (int r = wave; r < KK; r += 8) {
    float x[4];
    float s = 0.0f;
#pragma unroll
    for (int q = 0; q < 4; ++q) { x[q] = Es[r * E_STRIDE + lane + q * 32]; s += x[q]; }
    float m = wave_sum(s) * (1.0f / 128.0f);
    float vs = 0.0f;
#pragma unroll
    for (int q = 0; q < 4; ++q) { float d = x[q] - m; vs += d * d; }
    float rstd = rsqrtf(wave_sum(vs) * (1.0f / 128.0f) + LN_EPS);
#pragma unroll
    for (int q = 0; q < 4; ++q) {
      int cc = lane + q * 32;
      outE[((long)nf * KK + r) * HD + cc] =
          (x[q] - m) * rstd * ln3g[cc] + ln3b[cc];
    }
  }
}

// ---------------------------------------------------------------------------
extern "C" void kernel_launch(void* const* d_in, const int* in_sizes, int n_in,
                              void* d_out, int out_size, void* d_ws, size_t ws_size,
                              hipStream_t stream) {
  const float* hV    = (const float*)d_in[0];
  const float* hE    = (const float*)d_in[1];
  const int*   Eidx  = (const int*)d_in[2];
  const float* maskV = (const float*)d_in[3];
  const float* maskA = (const float*)d_in[4];
  const float* W1_w  = (const float*)d_in[5];
  const float* W1_b  = (const float*)d_in[6];
  const float* W2_w  = (const float*)d_in[7];
  const float* W2_b  = (const float*)d_in[8];
  const float* W3_w  = (const float*)d_in[9];
  const float* W3_b  = (const float*)d_in[10];
  const float* W11_w = (const float*)d_in[11];
  const float* W11_b = (const float*)d_in[12];
  const float* W12_w = (const float*)d_in[13];
  const float* W12_b = (const float*)d_in[14];
  const float* W13_w = (const float*)d_in[15];
  const float* W13_b = (const float*)d_in[16];
  const float* Win_w = (const float*)d_in[17];
  const float* Win_b = (const float*)d_in[18];
  const float* Wout_w= (const float*)d_in[19];
  const float* Wout_b= (const float*)d_in[20];
  const float* ln1g  = (const float*)d_in[21];
  const float* ln1b  = (const float*)d_in[22];
  const float* ln2g  = (const float*)d_in[23];
  const float* ln2b  = (const float*)d_in[24];
  const float* ln3g  = (const float*)d_in[25];
  const float* ln3b  = (const float*)d_in[26];

  char* ws = (char*)d_ws;
  unsigned short* W1p   = (unsigned short*)(ws + 0);       // 384*128*2  = 98304
  unsigned short* W2p   = (unsigned short*)(ws + 98304);   // 128*128*2  = 32768
  unsigned short* W3p   = (unsigned short*)(ws + 131072);
  unsigned short* W11p  = (unsigned short*)(ws + 163840);
  unsigned short* W12p  = (unsigned short*)(ws + 262144);
  unsigned short* W13p  = (unsigned short*)(ws + 294912);
  unsigned short* Winp  = (unsigned short*)(ws + 327680);  // 128*512*2 = 131072
  unsigned short* Woutp = (unsigned short*)(ws + 458752);  // 512*128*2 = 131072

  float* outV = (float*)d_out;                  // [B,N,H]  (also dh scratch)
  float* outE = (float*)d_out + (long)NNOD * HD;  // [B,N,K,H]

  // Repack weights into WMMA-B operand order (bf16)
  pack_w_kernel<<<192, 256, 0, stream>>>(W1_w, W1p, 384, 128);
  pack_w_kernel<<<64, 256, 0, stream>>>(W2_w, W2p, 128, 128);
  pack_w_kernel<<<64, 256, 0, stream>>>(W3_w, W3p, 128, 128);
  pack_w_kernel<<<192, 256, 0, stream>>>(W11_w, W11p, 384, 128);
  pack_w_kernel<<<64, 256, 0, stream>>>(W12_w, W12p, 128, 128);
  pack_w_kernel<<<64, 256, 0, stream>>>(W13_w, W13p, 128, 128);
  pack_w_kernel<<<256, 256, 0, stream>>>(Win_w, Winp, 128, 512);
  pack_w_kernel<<<256, 256, 0, stream>>>(Wout_w, Woutp, 512, 128);

  // Node messages -> dh (stored in outV region)
  node_msg_kernel<<<NNOD, 256, 0, stream>>>(hV, hE, Eidx, maskA, W1p, W2p, W3p,
                                            W1_b, W2_b, W3_b, outV);
  // LN1 + FFN + LN2 + mask -> h_Vn (overwrites outV region)
  node_ffn_kernel<<<NNOD / 16, 256, 0, stream>>>(hV, outV, Winp, Woutp, Win_b,
                                                 Wout_b, ln1g, ln1b, ln2g, ln2b,
                                                 maskV, outV);
  // Edge update -> h_En
  edge_upd_kernel<<<NNOD, 256, 0, stream>>>(outV, hE, Eidx, W11p, W12p, W13p,
                                            W11_b, W12_b, W13_b, ln3g, ln3b,
                                            outE);
}
